// ListaLayer_41102837022963
// MI455X (gfx1250) — matrix operand order repaired and checked by reading, not compiled, not run on hardware
//
#include <hip/hip_runtime.h>

#define LISTA_EPS 1e-7f

#if __has_builtin(__builtin_amdgcn_global_load_async_to_lds_b128)
#define USE_ASYNC_COPY 1
#else
#define USE_ASYNC_COPY 0
#endif

typedef __attribute__((ext_vector_type(16))) __bf16 v16bf;
typedef __attribute__((ext_vector_type(8)))  __bf16 v8bf;
typedef __attribute__((ext_vector_type(8)))  float  v8f;

union BF16x16 { v16bf v; v8bf h[2]; };

__device__ __forceinline__ v16bf load_frag(const __bf16* row, int o0, int o1) {
    BF16x16 u;
    u.h[0] = *(const v8bf*)(row + o0);
    u.h[1] = *(const v8bf*)(row + o1);
    return u.v;
}

__device__ __forceinline__ float soft_thresh(float z) {
    float a = fabsf(z) - 1.0f;
    a = a > 0.0f ? a : 0.0f;
    return copysignf(a, z);   // sign(z)*relu(|z|-1)
}

#if USE_ASYNC_COPY
// Builtin signature (from compiler diagnostic): param0 is
// 'int __vector(4) __device__(AS1) *', so pass typed int4 pointers.
typedef int v4i_vs __attribute__((vector_size(16)));
typedef __attribute__((address_space(1))) v4i_vs* gas_b128_ptr;
typedef __attribute__((address_space(3))) v4i_vs* las_b128_ptr;

__device__ __forceinline__ void async_copy_b128(void* lds_dst, const void* gsrc) {
    __builtin_amdgcn_global_load_async_to_lds_b128(
        (gas_b128_ptr)gsrc, (las_b128_ptr)lds_dst, 0, 0);
}
__device__ __forceinline__ void wait_async_zero() {
#if __has_builtin(__builtin_amdgcn_s_wait_asynccnt)
    __builtin_amdgcn_s_wait_asynccnt(0);
#else
    asm volatile("s_wait_asynccnt 0x0" ::: "memory");
#endif
}
#endif

// ---------------- prep kernels: fold th scalings, transpose B operands, bf16 ----

__global__ void conv_y_kernel(const float* __restrict__ y, __bf16* __restrict__ ybf) {
    int idx = blockIdx.x * 256 + threadIdx.x;
    ybf[idx] = (__bf16)y[idx];
}

// WT[n][k] = W[k][n] / th[n]   (W is [1024][2048], WT is [2048][1024])
__global__ void prep_wt_kernel(const float* __restrict__ W, const float* __restrict__ Theta,
                               __bf16* __restrict__ WT) {
    int idx = blockIdx.x * 256 + threadIdx.x;  // over 1024*2048, read-coalesced
    int k = idx >> 11;
    int n = idx & 2047;
    float th = fmaxf(Theta[n], 0.0f) + LISTA_EPS;
    WT[n * 1024 + k] = (__bf16)(W[idx] / th);
}

// ST[n][j] = th[j] * S[j][n] / th[n]   (S is [2048][2048], ST is [2048][2048])
__global__ void prep_st_kernel(const float* __restrict__ S, const float* __restrict__ Theta,
                               __bf16* __restrict__ ST) {
    int idx = blockIdx.x * 256 + threadIdx.x;  // over 2048*2048, read-coalesced
    int j = idx >> 11;
    int n = idx & 2047;
    float thj = fmaxf(Theta[j], 0.0f) + LISTA_EPS;
    float thn = fmaxf(Theta[n], 0.0f) + LISTA_EPS;
    ST[n * 2048 + j] = (__bf16)(S[idx] * thj / thn);
}

// DT[i][j] = th[j] * Dx[j][i]   (Dx is [2048][1024], DT is [1024][2048])
__global__ void prep_dt_kernel(const float* __restrict__ Dx, const float* __restrict__ Theta,
                               __bf16* __restrict__ DT) {
    int idx = blockIdx.x * 256 + threadIdx.x;  // over 2048*1024, read-coalesced
    int j = idx >> 10;
    int i = idx & 1023;
    float thj = fmaxf(Theta[j], 0.0f) + LISTA_EPS;
    DT[i * 2048 + j] = (__bf16)(Dx[idx] * thj);
}

// ---------------- main WMMA GEMM -------------------------------------------------
// C[M][N] = A[M][K] @ Bt[N][K]^T (+ Cinit)   with per-mode epilogue:
//   MODE 0: outF = acc (z0),  outBF = soft(acc)      (first GEMM)
//   MODE 1: acc init = Cinit (z0), outBF = soft(acc) (iteration GEMM)
//   MODE 2: outF = acc                                (final GEMM)
// 128 threads = 4 wave32s, each wave owns a 64x64 tile (4x4 WMMA accumulators).
template <int MODE>
__global__ __launch_bounds__(128) void lista_gemm(
    const __bf16* __restrict__ A, const __bf16* __restrict__ Bt,
    const float* __restrict__ Cinit, float* __restrict__ outF,
    __bf16* __restrict__ outBF, int M, int N, int K)
{
    (void)M;
    constexpr int BM = 128, BN = 128, BK = 32, LK = 48;  // 96B LDS rows, 16B aligned
    __shared__ __bf16 sA[2][BM][LK];
    __shared__ __bf16 sB[2][BN][LK];

    const int t    = threadIdx.x;
    const int bm   = blockIdx.y * BM;
    const int bn   = blockIdx.x * BN;
    const int wave = t >> 5;            // 0..3
    const int lane = t & 31;
    const int wm   = (wave >> 1) * 64;  // 2x2 wave grid
    const int wn   = (wave & 1) * 64;
    const int lr   = lane & 15;
    const int h    = lane >> 4;

    // staging: 128 threads x 16B, 4 row-passes per operand per k-tile
    const int sr = t >> 2;        // 0..31
    const int sc = (t & 3) * 8;   // 0,8,16,24
    const __bf16* gA = A  + (size_t)bm * K;
    const __bf16* gB = Bt + (size_t)bn * K;

    v8f acc[4][4];
#pragma unroll
    for (int i = 0; i < 4; ++i)
#pragma unroll
        for (int j = 0; j < 4; ++j) {
            if constexpr (MODE == 1) {
                const float* cp = Cinit + (size_t)(bm + wm + 16 * i + 8 * h) * N
                                        + (bn + wn + 16 * j + lr);
#pragma unroll
                for (int r = 0; r < 8; ++r) acc[i][j][r] = cp[(size_t)r * N];
            } else {
                acc[i][j] = (v8f){0.f, 0.f, 0.f, 0.f, 0.f, 0.f, 0.f, 0.f};
            }
        }

    // ---- stage k-tile 0 into buffer 0 ----
#if USE_ASYNC_COPY
#pragma unroll
    for (int p = 0; p < 4; ++p) {
        async_copy_b128(&sA[0][sr + 32 * p][sc], gA + (size_t)(sr + 32 * p) * K + sc);
        async_copy_b128(&sB[0][sr + 32 * p][sc], gB + (size_t)(sr + 32 * p) * K + sc);
    }
    wait_async_zero();
#else
#pragma unroll
    for (int p = 0; p < 4; ++p) {
        *(uint4*)&sA[0][sr + 32 * p][sc] = *(const uint4*)(gA + (size_t)(sr + 32 * p) * K + sc);
        *(uint4*)&sB[0][sr + 32 * p][sc] = *(const uint4*)(gB + (size_t)(sr + 32 * p) * K + sc);
    }
#endif
    __syncthreads();

    const int nk = K / BK;
    int cur = 0;
    for (int kt = 0; kt < nk; ++kt) {
        const bool more = (kt + 1 < nk);
        const size_t ko = (size_t)(kt + 1) * BK;

#if USE_ASYNC_COPY
        if (more) {  // issue next tile's async copies; they complete during the WMMAs
#pragma unroll
            for (int p = 0; p < 4; ++p) {
                async_copy_b128(&sA[cur ^ 1][sr + 32 * p][sc],
                                gA + (size_t)(sr + 32 * p) * K + ko + sc);
                async_copy_b128(&sB[cur ^ 1][sr + 32 * p][sc],
                                gB + (size_t)(sr + 32 * p) * K + ko + sc);
            }
        }
#else
        uint4 ra[4], rb[4];
        if (more) {
#pragma unroll
            for (int p = 0; p < 4; ++p) {
                ra[p] = *(const uint4*)(gA + (size_t)(sr + 32 * p) * K + ko + sc);
                rb[p] = *(const uint4*)(gB + (size_t)(sr + 32 * p) * K + ko + sc);
            }
        }
#endif

        // fragments per CDNA5 16-bit WMMA VGPR layout:
        //  A 16x32: lane holds K[8h..8h+7] then K[16+8h..16+8h+7]
        //  B 32x16 (Bt row-major over K): lane holds K[8h..8h+7] then K[16+8h..16+8h+7]
        v16bf af[4], bfr[4];
#pragma unroll
        for (int i = 0; i < 4; ++i)
            af[i] = load_frag(&sA[cur][wm + 16 * i + lr][0], 8 * h, 16 + 8 * h);
#pragma unroll
        for (int j = 0; j < 4; ++j)
            bfr[j] = load_frag(&sB[cur][wn + 16 * j + lr][0], 8 * h, 16 + 8 * h);

#pragma unroll
        for (int i = 0; i < 4; ++i)
#pragma unroll
            for (int j = 0; j < 4; ++j)
                acc[i][j] = __builtin_amdgcn_wmma_f32_16x16x32_bf16(
                    false, af[i], false, bfr[j], (short)0, acc[i][j], false, false);

#if !USE_ASYNC_COPY
        if (more) {
#pragma unroll
            for (int p = 0; p < 4; ++p) {
                *(uint4*)&sA[cur ^ 1][sr + 32 * p][sc] = ra[p];
                *(uint4*)&sB[cur ^ 1][sr + 32 * p][sc] = rb[p];
            }
        }
#endif
#if USE_ASYNC_COPY
        wait_async_zero();   // my async writes (next tile) have landed in LDS
#endif
        __syncthreads();     // everyone's writes visible; reads of old buffer done
        cur ^= 1;
    }

    // epilogue (C/D layout: element r -> row = 16*i + r + 8h, col = 16*j + lr)
#pragma unroll
    for (int i = 0; i < 4; ++i)
#pragma unroll
        for (int j = 0; j < 4; ++j) {
            const int row0 = bm + wm + 16 * i + 8 * h;
            const int col  = bn + wn + 16 * j + lr;
#pragma unroll
            for (int r = 0; r < 8; ++r) {
                const size_t idx = (size_t)(row0 + r) * N + col;
                const float v = acc[i][j][r];
                if constexpr (MODE == 2) {
                    outF[idx] = v;
                } else {
                    if constexpr (MODE == 0) outF[idx] = v;
                    outBF[idx] = (__bf16)soft_thresh(v);
                }
            }
        }
}

// ---------------- host-side orchestration ---------------------------------------

extern "C" void kernel_launch(void* const* d_in, const int* in_sizes, int n_in,
                              void* d_out, int out_size, void* d_ws, size_t ws_size,
                              hipStream_t stream) {
    (void)in_sizes; (void)n_in; (void)out_size; (void)ws_size;
    const float* y     = (const float*)d_in[0];
    const float* W     = (const float*)d_in[1];
    const float* Theta = (const float*)d_in[2];
    const float* S     = (const float*)d_in[3];
    const float* Dx    = (const float*)d_in[4];

    const int B = 16384, Din = 1024, Dd = 2048, STEPS = 16;

    char* ws = (char*)d_ws;
    size_t off = 0;
    auto take = [&](size_t bytes) {
        char* p = ws + off;
        off += (bytes + 255) & ~(size_t)255;
        return p;
    };
    __bf16* ybf = (__bf16*)take((size_t)B * Din * 2);
    __bf16* WT  = (__bf16*)take((size_t)Dd * Din * 2);
    __bf16* ST  = (__bf16*)take((size_t)Dd * Dd * 2);
    __bf16* DT  = (__bf16*)take((size_t)Din * Dd * 2);
    float*  z0  = (float*) take((size_t)B * Dd * 4);
    __bf16* a0  = (__bf16*)take((size_t)B * Dd * 2);
    __bf16* a1  = (__bf16*)take((size_t)B * Dd * 2);

    conv_y_kernel<<<(B * Din) / 256, 256, 0, stream>>>(y, ybf);
    prep_wt_kernel<<<(Din * Dd) / 256, 256, 0, stream>>>(W, Theta, WT);
    prep_st_kernel<<<(Dd * Dd) / 256, 256, 0, stream>>>(S, Theta, ST);
    prep_dt_kernel<<<(Dd * Din) / 256, 256, 0, stream>>>(Dx, Theta, DT);

    dim3 blk(128);
    dim3 g0(Dd / 128, B / 128);
    // z0 = y @ W'; a0 = soft(z0)
    lista_gemm<0><<<g0, blk, 0, stream>>>(ybf, WT, nullptr, z0, a0, B, Dd, Din);

    // 16x: z = z0 + a @ S'; a = soft(z)
    __bf16* acur = a0; __bf16* anext = a1;
    for (int t = 0; t < STEPS; ++t) {
        lista_gemm<1><<<g0, blk, 0, stream>>>(acur, ST, z0, nullptr, anext, B, Dd, Dd);
        __bf16* tmp = acur; acur = anext; anext = tmp;
    }

    // out = a @ Dx'
    dim3 g2(Din / 128, B / 128);
    lista_gemm<2><<<g2, blk, 0, stream>>>(acur, DT, nullptr, (float*)d_out, nullptr, B, Din, Dd);
}
